// HungarianMatcher_18124761989207
// MI455X (gfx1250) — matrix operand order repaired and checked by reading, not compiled, not run on hardware
//
#include <hip/hip_runtime.h>
#include <hip/hip_bf16.h>
#include <math.h>

typedef __attribute__((ext_vector_type(16))) _Float16 v16h;
typedef __attribute__((ext_vector_type(8)))  float    v8f;

#define Bn    4
#define Qn    100
#define CLS   81
#define Tn    20
#define HGn   512
#define WGn   512
#define HWn   65536          // 256*256
#define KCH   2048           // HWn / 32
#define QT    7              // ceil(100/16)
#define QPAD  112            // QT*16
#define NT    2              // ceil(20/16)
#define TPAD  32             // NT*16
#define KSPLIT 32
#define KC_PER (KCH / KSPLIT)

#define LOG2E 1.4426950408889634f
#define LN2   0.6931471805599453f

union HF16 { v16h v; _Float16 h[16]; };
union F8   { v8f  v; float    f[8];  };

// ---------------------------------------------------------------------------
// Zero the float accumulator region of the workspace
// ---------------------------------------------------------------------------
__global__ void zero_ws(float* __restrict__ p, int n) {
    int i = blockIdx.x * blockDim.x + threadIdx.x;
    if (i < n) p[i] = 0.0f;
}

// ---------------------------------------------------------------------------
// Downsample tgt_masks (nearest, 512->256) and pack into WMMA B-fragments.
// B (32x16 f16) packing used here (same K order as A): lanes 0-15 hold N=lane, K=0..15; lanes 16-31 hold
// N=lane-16, K=16..31 (one K-row striped across lanes within a VGPR).
// One wave per (b, kchunk, ntile); lane writes one v16h (32B, coalesced).
// Source pixels are stride-2 in x, so each aligned float4 load yields two
// downsampled values (.x and .z). A 16-K run never crosses a source row.
// ---------------------------------------------------------------------------
__global__ __launch_bounds__(32) void pack_tgt(const float* __restrict__ tgt,
                                               _Float16* __restrict__ tmB,
                                               float* __restrict__ tmsum) {
    int x    = blockIdx.x;
    int nt   = x % NT;
    int kc   = (x / NT) % KCH;
    int b    = x / (NT * KCH);
    int lane = threadIdx.x;

    int t  = nt * 16 + (lane & 15);
    int kb = kc * 32 + ((lane < 16) ? 0 : 16);   // first K of this lane's run
    int y  = kb >> 8;                            // dst row (same for all 16 K)
    int xb = kb & 255;                           // dst col of first K

    const float* row = tgt + ((size_t)b * Tn + (t < Tn ? t : 0)) * (size_t)HGn * WGn
                           + (size_t)(2 * y) * WGn + 2 * xb;

    HF16 frag;
    float s = 0.0f;
    if (t < Tn) {
        const float4* r4 = (const float4*)row;   // 16B aligned (2*xb % 32 == 0)
#pragma unroll
        for (int j = 0; j < 8; ++j) {
            float4 f = r4[j];                    // src cols 2xb+4j .. +3
            frag.h[2 * j]     = (_Float16)f.x;   // dst col xb+2j
            frag.h[2 * j + 1] = (_Float16)f.z;   // dst col xb+2j+1
            s += f.x + f.z;
        }
    } else {
#pragma unroll
        for (int j = 0; j < 16; ++j) frag.h[j] = (_Float16)0.0f;
    }

    v16h* out = (v16h*)tmB;
    out[(((size_t)b * KCH + kc) * NT + nt) * 32 + lane] = frag.v;
    if (t < Tn) atomicAdd(&tmsum[b * TPAD + t], s);
}

// ---------------------------------------------------------------------------
// Core GEMM: S1[b,q,t] += pm . tm   and   S2[b,q,t] += sigmoid(pm) . tm
// plus per-row sums of softplus(pm) and sigmoid(pm).
// One wave per (b, q-tile, k-split); K loop over chunks of 32 with
// V_WMMA_F32_16X16X32_F16, f32 accumulation, atomic reduction across splits.
// A (16x32 f16) packing used here (same K order as B): lanes 0-15: M=lane, halves=[K0..15];
// lanes 16-31: M=lane-16, halves=[K16..31].
// Elementwise math uses raw v_exp_f32/v_log_f32/v_rcp_f32 (3 trans + ~8 VALU
// per element) to stay under the 8.2us HBM roofline instead of libm expf.
// ---------------------------------------------------------------------------
__global__ __launch_bounds__(32) void cost_gemm(const float* __restrict__ pm,
                                                const _Float16* __restrict__ tmB,
                                                float* __restrict__ S1,
                                                float* __restrict__ S2,
                                                float* __restrict__ spsum,
                                                float* __restrict__ sgsum) {
    int x    = blockIdx.x;
    int ks   = x % KSPLIT;
    int qt   = (x / KSPLIT) % QT;
    int b    = x / (KSPLIT * QT);
    int lane = threadIdx.x;

    int q0   = qt * 16;
    int qrow = q0 + (lane & 15);
    int qc   = (qrow < Qn) ? qrow : (Qn - 1);    // clamp load row (pads unused)
    const float* prow = pm + ((size_t)b * Qn + qc) * (size_t)HWn;
    const v16h* tb = (const v16h*)tmB;

    v8f accP0 = {}, accP1 = {}, accS0 = {}, accS1 = {};
    float sp_acc = 0.0f, sg_acc = 0.0f;

    int kc0 = ks * KC_PER;
    for (int kc = kc0; kc < kc0 + KC_PER; ++kc) {
        int kb = kc * 32 + ((lane < 16) ? 0 : 16);
        const float4* p4 = (const float4*)(prow + kb);
        float4 f0 = p4[0];      // K = kb+0..3
        float4 f1 = p4[1];      // K = kb+4..7
        float4 f2 = p4[2];      // K = kb+8..11
        float4 f3 = p4[3];      // K = kb+12..15
        __builtin_prefetch(prow + kb + 1024, 0, 0);   // global_prefetch_b8

        float xs[16] = { f0.x, f0.y, f0.z, f0.w, f1.x, f1.y, f1.z, f1.w,
                         f2.x, f2.y, f2.z, f2.w, f3.x, f3.y, f3.z, f3.w };

        HF16 ap, ag;
#pragma unroll
        for (int j = 0; j < 16; ++j) {
            float v = xs[j];
            // e = exp(-|v|) via v_exp_f32; shared by sigmoid and softplus
            float e  = __builtin_amdgcn_exp2f(-fabsf(v) * LOG2E);
            float r  = __builtin_amdgcn_rcpf(1.0f + e);        // sigmoid(|v|)
            float sg = (v >= 0.0f) ? r : e * r;                // sigmoid(v)
            // softplus(v) = max(v,0) + ln(1+e) ; v_log_f32 is log2
            float sp = fmaxf(v, 0.0f) + LN2 * __builtin_amdgcn_logf(1.0f + e);
            sp_acc  += sp;
            sg_acc  += sg;
            ap.h[j]  = (_Float16)v;
            ag.h[j]  = (_Float16)sg;
        }

        size_t fb = (((size_t)b * KCH + kc) * NT) * 32 + lane;
        v16h b0 = tb[fb];          // N-tile 0 fragment
        v16h b1 = tb[fb + 32];     // N-tile 1 fragment

        accP0 = __builtin_amdgcn_wmma_f32_16x16x32_f16(false, ap.v, false, b0,
                                                       (short)0, accP0, false, false);
        accP1 = __builtin_amdgcn_wmma_f32_16x16x32_f16(false, ap.v, false, b1,
                                                       (short)0, accP1, false, false);
        accS0 = __builtin_amdgcn_wmma_f32_16x16x32_f16(false, ag.v, false, b0,
                                                       (short)0, accS0, false, false);
        accS1 = __builtin_amdgcn_wmma_f32_16x16x32_f16(false, ag.v, false, b1,
                                                       (short)0, accS1, false, false);
    }

    // Row partial sums (lanes l and l+16 both hold partials of row qrow).
    atomicAdd(&spsum[b * QPAD + qrow], sp_acc);
    atomicAdd(&sgsum[b * QPAD + qrow], sg_acc);

    // C/D layout: VGPR r, lane l<16 -> (M=r, N=l); lane l>=16 -> (M=8+r, N=l-16)
    F8 uP0, uP1, uS0, uS1;
    uP0.v = accP0; uP1.v = accP1; uS0.v = accS0; uS1.v = accS1;
    int tcol = (lane & 15);
#pragma unroll
    for (int r = 0; r < 8; ++r) {
        int qr = q0 + ((lane < 16) ? r : (8 + r));
        size_t base = ((size_t)b * QPAD + qr) * TPAD;
        atomicAdd(&S1[base + tcol],      uP0.f[r]);
        atomicAdd(&S1[base + 16 + tcol], uP1.f[r]);
        atomicAdd(&S2[base + tcol],      uS0.f[r]);
        atomicAdd(&S2[base + 16 + tcol], uS1.f[r]);
    }
}

// ---------------------------------------------------------------------------
// Combine: C = 2*cost_class + 5*cost_mask + 5*cost_dice, nan_to_num.
// cost_mask = (sum softplus(pm) - pm.tm) / HW
// cost_dice = 1 - (2*sig.tm + 1) / (sum sig + sum tm + 1)
// cost_class = -softmax(logits)[label]
// ---------------------------------------------------------------------------
__global__ void finalize_cost(const float* __restrict__ logits,
                              const int* __restrict__ labels,
                              const float* __restrict__ S1,
                              const float* __restrict__ S2,
                              const float* __restrict__ spsum,
                              const float* __restrict__ sgsum,
                              const float* __restrict__ tmsum,
                              float* __restrict__ out) {
    int idx = blockIdx.x * blockDim.x + threadIdx.x;
    if (idx >= Bn * Qn * Tn) return;
    int t = idx % Tn;
    int q = (idx / Tn) % Qn;
    int b = idx / (Tn * Qn);

    size_t base = ((size_t)b * QPAD + q) * TPAD + t;
    float s1 = S1[base];
    float s2 = S2[base];
    float sp = spsum[b * QPAD + q];
    float sg = sgsum[b * QPAD + q];
    float ts = tmsum[b * TPAD + t];

    float cmask = (sp - s1) / (float)HWn;
    float cdice = 1.0f - (2.0f * s2 + 1.0f) / (sg + ts + 1.0f);

    const float* lr = logits + ((size_t)b * Qn + q) * CLS;
    int lbl = labels[b * Tn + t];
    if (lbl < 0) lbl = 0;
    if (lbl >= CLS) lbl = CLS - 1;
    float mx = lr[0];
    for (int c = 1; c < CLS; ++c) mx = fmaxf(mx, lr[c]);
    float den = 0.0f;
    for (int c = 0; c < CLS; ++c) den += __builtin_amdgcn_exp2f((lr[c] - mx) * LOG2E);
    float prob = __builtin_amdgcn_exp2f((lr[lbl] - mx) * LOG2E) / den;
    float cclass = -prob;

    float cost = 2.0f * cclass + 5.0f * cmask + 5.0f * cdice;
    if (isnan(cost)) cost = 0.0f;
    else if (isinf(cost)) cost = (cost > 0.0f) ? 3.402823466e+38f : -3.402823466e+38f;

    out[((size_t)b * Qn + q) * Tn + t] = cost;
}

// ---------------------------------------------------------------------------
// Shortest-augmenting-path LSA on C^T (rows=T=20, cols=Q=100), one block per
// batch, serial on lane 0 (problem is tiny). Then sort pairs by query index.
// ---------------------------------------------------------------------------
__global__ __launch_bounds__(32) void lsa_kernel(const float* __restrict__ Cmat,
                                                 int* __restrict__ pidx,
                                                 int* __restrict__ tidx) {
    if (threadIdx.x != 0) return;
    int b = blockIdx.x;
    const float* Cb = Cmat + (size_t)b * Qn * Tn;   // [Q][T]

    double u[Tn], vv[Qn], shortest[Qn];
    int path[Qn], row4col[Qn], col4row[Tn];
    bool SR[Tn], SC[Qn];

    for (int i = 0; i < Tn; ++i) { u[i] = 0.0; col4row[i] = -1; }
    for (int j = 0; j < Qn; ++j) { vv[j] = 0.0; row4col[j] = -1; }

    for (int cur = 0; cur < Tn; ++cur) {
        for (int j = 0; j < Qn; ++j) { shortest[j] = 1e300; path[j] = -1; SC[j] = false; }
        for (int i = 0; i < Tn; ++i) SR[i] = false;
        int i = cur;
        double minVal = 0.0;
        int sink = -1;
        while (sink == -1) {
            SR[i] = true;
            double best = 1e301;
            int jloc = -1;
            for (int j = 0; j < Qn; ++j) {
                if (SC[j]) continue;
                double d = minVal + (double)Cb[(size_t)j * Tn + i] - u[i] - vv[j];
                if (d < shortest[j]) { shortest[j] = d; path[j] = i; }
                if (shortest[j] < best) { best = shortest[j]; jloc = j; }
            }
            minVal = best;
            SC[jloc] = true;
            if (row4col[jloc] == -1) sink = jloc;
            else i = row4col[jloc];
        }
        u[cur] += minVal;
        for (int r = 0; r < Tn; ++r)
            if (SR[r] && r != cur) u[r] += minVal - shortest[col4row[r]];
        for (int j = 0; j < Qn; ++j)
            if (SC[j]) vv[j] -= minVal - shortest[j];
        int j = sink;
        while (true) {
            int ii = path[j];
            row4col[j] = ii;
            int nxt = col4row[ii];
            col4row[ii] = j;
            j = nxt;
            if (ii == cur) break;
        }
    }

    int qs[Tn], ts[Tn];
    for (int t = 0; t < Tn; ++t) { qs[t] = col4row[t]; ts[t] = t; }
    for (int a = 0; a < Tn - 1; ++a) {            // sort by query ascending
        int m = a;
        for (int c = a + 1; c < Tn; ++c) if (qs[c] < qs[m]) m = c;
        int tq = qs[a]; qs[a] = qs[m]; qs[m] = tq;
        int tt = ts[a]; ts[a] = ts[m]; ts[m] = tt;
    }
    for (int t = 0; t < Tn; ++t) {
        pidx[b * Tn + t] = qs[t];
        tidx[b * Tn + t] = ts[t];
    }
}

// ---------------------------------------------------------------------------
extern "C" void kernel_launch(void* const* d_in, const int* in_sizes, int n_in,
                              void* d_out, int out_size, void* d_ws, size_t ws_size,
                              hipStream_t stream) {
    const float* pred_logits = (const float*)d_in[0];   // [4,100,81]
    const float* pred_masks  = (const float*)d_in[1];   // [4,100,256,256]
    const float* tgt_masks   = (const float*)d_in[2];   // [4,20,512,512]
    const int*   tgt_labels  = (const int*)d_in[3];     // [4,20]

    // Workspace layout
    _Float16* tmB = (_Float16*)d_ws;
    size_t tmB_halves = (size_t)Bn * KCH * NT * 32 * 16;      // 8.4M halves
    float* S1    = (float*)(tmB + tmB_halves);
    float* S2    = S1 + (size_t)Bn * QPAD * TPAD;
    float* spsum = S2 + (size_t)Bn * QPAD * TPAD;
    float* sgsum = spsum + (size_t)Bn * QPAD;
    float* tmsum = sgsum + (size_t)Bn * QPAD;
    int nz = 2 * Bn * QPAD * TPAD + 2 * Bn * QPAD + Bn * TPAD;

    // Output layout: C [B*Q*T] f32, then pred_idx [B*T] i32, tgt_idx [B*T] i32
    float* Cout = (float*)d_out;
    int* pidx = (int*)(Cout + (size_t)Bn * Qn * Tn);
    int* tidx = pidx + Bn * Tn;

    zero_ws<<<(nz + 255) / 256, 256, 0, stream>>>(S1, nz);
    pack_tgt<<<Bn * KCH * NT, 32, 0, stream>>>(tgt_masks, tmB, tmsum);
    cost_gemm<<<Bn * QT * KSPLIT, 32, 0, stream>>>(pred_masks, tmB,
                                                   S1, S2, spsum, sgsum);
    finalize_cost<<<(Bn * Qn * Tn + 255) / 256, 256, 0, stream>>>(
        pred_logits, tgt_labels, S1, S2, spsum, sgsum, tmsum, Cout);
    lsa_kernel<<<Bn, 32, 0, stream>>>(Cout, pidx, tidx);
}
